// CTRNN_MD_35682588295218
// MI455X (gfx1250) — matrix operand order repaired and checked by reading, not compiled, not run on hardware
//
#include <hip/hip_runtime.h>

// ---------------------------------------------------------------------------
// CTRNN-MD on MI455X (gfx1250).
// Recurrence is independent per batch row: 16 persistent workgroups, each owns
// 16 batch rows for all T=1024 steps; h state in LDS (f32 + bf16 shadow),
// md state in wave-7 registers. All GEMMs via v_wmma_f32_16x16x32_bf16.
// Keep-side x2h/h2h GEMMs share one accumulator; biases folded into C init.
// x_t tile staged with GLOBAL_LOAD_ASYNC_TO_LDS (ASYNCcnt) when available.
// ---------------------------------------------------------------------------

#define T_STEPS 1024
#define BATCH   256
#define IDIM    128
#define HDIM    512
#define MDIM    16
#define ODIM    32
#define MD_KEEP 0.7f

// bf16 workspace layout (units: unsigned short elements)
#define WS_H2H   0         // 1024 x 512
#define WS_X2H   524288    // 1024 x 128
#define WS_H2MD  655360    //   16 x 512
#define WS_X2MD  663552    //   16 x 128
#define WS_MG    665600    //   16 x 512
#define WS_H2R   673792    //   32 x 512
#define WS_TOTAL 690176

#define AS1 __attribute__((address_space(1)))
#define AS3 __attribute__((address_space(3)))

#if defined(__has_builtin)
#if __has_builtin(__builtin_amdgcn_global_load_async_to_lds_b128)
#define HAVE_ASYNC_LDS 1
#endif
#endif

typedef __attribute__((ext_vector_type(16))) __bf16 v16bf;
typedef __attribute__((ext_vector_type(8)))  float  v8f;
typedef __attribute__((ext_vector_type(4)))  int    v4i;

union Frag {
    v16bf          v;
    uint4          q[2];
    unsigned short u[16];
};

__device__ __forceinline__ unsigned short f2bf(float f) {
    union { float f; unsigned int u; } a;
    a.f = f;
    unsigned int r = a.u + 0x7FFFu + ((a.u >> 16) & 1u); // round-to-nearest-even
    return (unsigned short)(r >> 16);
}

__device__ __forceinline__ float sigm(float x) { return 1.0f / (1.0f + __expf(-x)); }

// Deterministic hash-based Gumbel noise (stand-in for jax threefry key 42).
__device__ __forceinline__ float gumbel_noise(unsigned t, unsigned b, unsigned k) {
    unsigned s = t * 2654435761u ^ (b * 40503u + 0x9E3779B9u) ^ (k * 668265263u);
    s ^= s >> 16; s *= 0x7feb352du;
    s ^= s >> 15; s *= 0x846ca68bu;
    s ^= s >> 16;
    float u = ((float)(s >> 8) + 1.0f) * (1.0f / 16777216.0f); // (0,1]
    return -__logf(-__logf(u));
}

__device__ __forceinline__ void wait_asynccnt0() {
#if defined(__has_builtin)
#if __has_builtin(__builtin_amdgcn_s_wait_asynccnt)
    __builtin_amdgcn_s_wait_asynccnt(0);
    return;
#endif
#endif
    asm volatile("s_wait_asynccnt 0" ::: "memory");
}

__device__ __forceinline__ v8f wmma_bf16(const Frag& a, const Frag& b, v8f c) {
    return __builtin_amdgcn_wmma_f32_16x16x32_bf16(
        false, a.v, false, b.v, (short)0, c, false, false);
}

// A-fragment from a row-major bf16 LDS tile: lane lm = row m, half picks the
// documented K split (0..7,16..23 vs 8..15,24..31). Two ds_load_b128.
__device__ __forceinline__ Frag loadA16(const unsigned short* __restrict__ base,
                                        int ld, int lm, int k0, int half) {
    Frag a;
    const unsigned short* r = base + lm * ld;
    a.q[0] = *(const uint4*)(r + k0 + (half ? 8 : 0));
    a.q[1] = *(const uint4*)(r + k0 + 16 + (half ? 8 : 0));
    return a;
}

// B-fragment: B[k][n] = W[row0+n][k]; lane n (=lm), half picks the
// documented K split (0..7,16..23 vs 8..15,24..31), matching loadA16.
__device__ __forceinline__ Frag loadB(const unsigned short* __restrict__ w,
                                      int row, int ldk, int k0, int half) {
    Frag b;
    const unsigned short* p = w + (size_t)row * ldk + k0 + half * 8;
    b.q[0] = *(const uint4*)(p);
    b.q[1] = *(const uint4*)(p + 16);
    return b;
}

__device__ __forceinline__ v8f bcast8(float x) {
    v8f r;
    #pragma unroll
    for (int v = 0; v < 8; ++v) r[v] = x;
    return r;
}

// ---------------------------------------------------------------------------
// prep: fp32 weights -> bf16 workspace (one pass, fully parallel)
// ---------------------------------------------------------------------------
__global__ void prep_weights(const float* __restrict__ x2h, const float* __restrict__ h2h,
                             const float* __restrict__ h2md, const float* __restrict__ x2md,
                             const float* __restrict__ mg, const float* __restrict__ h2r,
                             unsigned short* __restrict__ ws) {
    int i = blockIdx.x * blockDim.x + threadIdx.x;
    if (i < 524288)      ws[WS_H2H  + i]            = f2bf(h2h[i]);
    else if (i < 655360) ws[WS_X2H  + (i - 524288)] = f2bf(x2h[i - 524288]);
    else if (i < 663552) ws[WS_H2MD + (i - 655360)] = f2bf(h2md[i - 655360]);
    else if (i < 665600) ws[WS_X2MD + (i - 663552)] = f2bf(x2md[i - 663552]);
    else if (i < 673792) ws[WS_MG   + (i - 665600)] = f2bf(mg[i - 665600]);
    else if (i < 690176) ws[WS_H2R  + (i - 673792)] = f2bf(h2r[i - 673792]);
}

// ---------------------------------------------------------------------------
// persistent recurrence kernel: 16 blocks x 256 threads (8 wave32s)
// ---------------------------------------------------------------------------
__global__ __launch_bounds__(256, 1)
void ctrnn_md_kernel(const float* __restrict__ x,
                     const float* __restrict__ x2h_b,
                     const float* __restrict__ h2h_b,
                     const float* __restrict__ h2md_b,
                     const float* __restrict__ x2md_b,
                     const float* __restrict__ h2r_b,
                     const unsigned short* __restrict__ ws,
                     float* __restrict__ out) {
    __shared__ float sh_h32[16][HDIM];                              // h state (f32)
    __shared__ __align__(16) unsigned short sh_h16[16][HDIM];       // h state (bf16 shadow)
    __shared__ __align__(16) unsigned short sh_x16[16][IDIM];       // x_t tile (bf16)
    __shared__ __align__(16) float sh_gate[16][HDIM];               // md@mul_gates; first
                                                                    // 8KB doubles as x stage
    __shared__ __align__(16) unsigned short sh_mgT[HDIM][16];       // mul_gates^T (bf16)
    __shared__ __align__(16) unsigned short sh_md16[16][MDIM];      // md (bf16)
    __shared__ float sh_hb[2 * HDIM];
    __shared__ float sh_xb[2 * HDIM];
    __shared__ float sh_mdb[MDIM];
    __shared__ float sh_rb[ODIM];

    const int tid  = threadIdx.x;
    const int wv   = tid >> 5;
    const int lane = tid & 31;
    const int lm   = lane & 15;   // m for A-frags, n for B-frags / C columns
    const int half = lane >> 4;
    const int M0   = blockIdx.x * 16;   // batch-row base for this WG

    const unsigned short* w_h2h  = ws + WS_H2H;
    const unsigned short* w_x2h  = ws + WS_X2H;
    const unsigned short* w_h2md = ws + WS_H2MD;
    const unsigned short* w_x2md = ws + WS_X2MD;
    const unsigned short* w_h2r  = ws + WS_H2R;

    // ---- one-time init ----
    for (int i = tid; i < 2 * HDIM; i += 256) { sh_hb[i] = h2h_b[i]; sh_xb[i] = x2h_b[i]; }
    if (tid < MDIM) sh_mdb[tid] = h2md_b[tid] + x2md_b[tid];
    if (tid < ODIM) sh_rb[tid]  = h2r_b[tid];
    for (int i = tid; i < 16 * HDIM; i += 256) {
        ((float*)sh_h32)[i] = 0.0f;
        ((unsigned short*)sh_h16)[i] = 0;
    }
    for (int j = tid; j < HDIM; j += 256) {
        #pragma unroll
        for (int k = 0; k < MDIM; ++k) sh_mgT[j][k] = ws[WS_MG + k * HDIM + j];
    }
    float mdreg[8];
    #pragma unroll
    for (int v = 0; v < 8; ++v) mdreg[v] = 0.0f;

    __syncthreads();

    const v8f zf = {0.f, 0.f, 0.f, 0.f, 0.f, 0.f, 0.f, 0.f};
    const uint4 zq = make_uint4(0u, 0u, 0u, 0u);

    for (int t = 0; t < T_STEPS; ++t) {
        // ---- phase 0: stage x_t tile into LDS as bf16 ----
        const float* xt = x + ((size_t)t * BATCH + M0) * IDIM;
#if defined(HAVE_ASYNC_LDS)
        {
            // async-copy 16x128 f32 (8KB) into dead sh_gate region, then each
            // thread converts exactly the chunks it staged (same-wave order,
            // ASYNCcnt==0 => data visible to issuing wave without a barrier).
            float* stage = &sh_gate[0][0];
            for (int c = tid; c < (16 * IDIM) / 4; c += 256) {
                __builtin_amdgcn_global_load_async_to_lds_b128(
                    (AS1 v4i*)(xt + c * 4), (AS3 v4i*)(stage + c * 4), 0, 0);
            }
            wait_asynccnt0();
            for (int c = tid; c < (16 * IDIM) / 4; c += 256) {
                #pragma unroll
                for (int j = 0; j < 4; ++j) {
                    const int e = c * 4 + j;
                    sh_x16[e >> 7][e & 127] = f2bf(stage[e]);
                }
            }
        }
#else
        for (int i = tid; i < 16 * IDIM; i += 256) {
            int m = i >> 7, k = i & 127;
            sh_x16[m][k] = f2bf(xt[m * IDIM + k]);
        }
#endif
        if (t + 1 < T_STEPS)
            __builtin_prefetch(x + ((size_t)(t + 1) * BATCH + M0) * IDIM + tid * 8, 0, 0);
        __syncthreads();

        // ---- phase 1: fused GEMMs.
        //   a_k[p]  = keep pre-activation = x@x2h_keep + h@h2h_keep + biases
        //   a_he[p] = ext_in  = h@h2h_ext + bias
        //   a_xe[p] = rec_in  = x@x2h_ext + bias   (gate-scaled later)
        v8f a_k[4], a_he[4], a_xe[4];
        #pragma unroll
        for (int p = 0; p < 4; ++p) {
            const int col = (wv * 4 + p) * 16 + lm;
            a_k[p]  = bcast8(sh_hb[col] + sh_xb[col]);
            a_he[p] = bcast8(sh_hb[HDIM + col]);
            a_xe[p] = bcast8(sh_xb[HDIM + col]);
        }

        for (int k0 = 0; k0 < IDIM; k0 += 32) {
            Frag A = loadA16(&sh_x16[0][0], IDIM, lm, k0, half);
            #pragma unroll
            for (int p = 0; p < 4; ++p) {
                const int jt = wv * 4 + p;
                Frag Bk = loadB(w_x2h, jt * 16 + lm, IDIM, k0, half);
                a_k[p]  = wmma_bf16(A, Bk, a_k[p]);
                Frag Be = loadB(w_x2h, HDIM + jt * 16 + lm, IDIM, k0, half);
                a_xe[p] = wmma_bf16(A, Be, a_xe[p]);
            }
        }
        for (int k0 = 0; k0 < HDIM; k0 += 32) {
            Frag A = loadA16(&sh_h16[0][0], HDIM, lm, k0, half);
            #pragma unroll
            for (int p = 0; p < 4; ++p) {
                const int jt = wv * 4 + p;
                Frag Bk = loadB(w_h2h, jt * 16 + lm, HDIM, k0, half);
                a_k[p]  = wmma_bf16(A, Bk, a_k[p]);
                Frag Be = loadB(w_h2h, HDIM + jt * 16 + lm, HDIM, k0, half);
                a_he[p] = wmma_bf16(A, Be, a_he[p]);
            }
        }

        // ---- phase 1b (wave 7): md path + gate = md @ mul_gates ----
        if (wv == 7) {
            v8f am = bcast8(sh_mdb[lm]);   // h2md_b + x2md_b folded into C
            for (int k0 = 0; k0 < HDIM; k0 += 32) {
                Frag A = loadA16(&sh_h16[0][0], HDIM, lm, k0, half);
                Frag B = loadB(w_h2md, lm, HDIM, k0, half);
                am = wmma_bf16(A, B, am);
            }
            for (int k0 = 0; k0 < IDIM; k0 += 32) {
                Frag A = loadA16(&sh_x16[0][0], IDIM, lm, k0, half);
                Frag B = loadB(w_x2md, lm, IDIM, k0, half);
                am = wmma_bf16(A, B, am);
            }
            #pragma unroll
            for (int v = 0; v < 8; ++v) {
                const int m = v + 8 * half;
                float lg = fmaxf(am[v], 0.0f) +
                           gumbel_noise((unsigned)t, (unsigned)(M0 + m), (unsigned)lm);
                float mx = lg;
                #pragma unroll
                for (int s = 1; s < 16; s <<= 1) mx = fmaxf(mx, __shfl_xor(mx, s, 32));
                float e = __expf(lg - mx);
                float sum = e;
                #pragma unroll
                for (int s = 1; s < 16; s <<= 1) sum += __shfl_xor(sum, s, 32);
                mdreg[v] = MD_KEEP * mdreg[v] + (1.0f - MD_KEEP) * (e / sum);
                sh_md16[m][lm] = f2bf(mdreg[v]);
            }
            asm volatile("s_wait_dscnt 0" ::: "memory");   // same-wave LDS RAW
            __builtin_amdgcn_wave_barrier();

            Frag Am;                                       // md as A, K padded 16->32
            Am.q[0] = *(const uint4*)&sh_md16[lm][half ? 8 : 0];
            Am.q[1] = zq;
            for (int nt = 0; nt < 32; ++nt) {
                Frag Bg;
                Bg.q[0] = *(const uint4*)&sh_mgT[nt * 16 + lm][half ? 8 : 0];
                Bg.q[1] = zq;
                v8f g = wmma_bf16(Am, Bg, zf);
                #pragma unroll
                for (int v = 0; v < 8; ++v) sh_gate[v + 8 * half][nt * 16 + lm] = g[v];
            }
        }
        __syncthreads();   // gate visible to all waves

        // ---- phase 2: elementwise recurrence update ----
        #pragma unroll
        for (int p = 0; p < 4; ++p) {
            const int col = (wv * 4 + p) * 16 + lm;
            #pragma unroll
            for (int v = 0; v < 8; ++v) {
                const int m    = v + 8 * half;
                const float keep = sigm(a_k[p][v]);
                const float rec  = a_xe[p][v] * sh_gate[m][col];
                const float hold = sh_h32[m][col];
                const float hnew = keep * hold + (1.0f - keep) * sigm(a_he[p][v] + rec);
                sh_h32[m][col] = hnew;
                sh_h16[m][col] = f2bf(hnew);
            }
        }
        __syncthreads();   // h_new (f32 + bf16) complete

        // ---- phase 3: out_t = relu(h_new @ h2r^T + b)  (waves 0..1) ----
        if (wv < 2) {
            v8f ao = bcast8(sh_rb[wv * 16 + lm]);
            for (int k0 = 0; k0 < HDIM; k0 += 32) {
                Frag A = loadA16(&sh_h16[0][0], HDIM, lm, k0, half);
                Frag B = loadB(w_h2r, wv * 16 + lm, HDIM, k0, half);
                ao = wmma_bf16(A, B, ao);
            }
            float* op = out + ((size_t)t * BATCH + M0) * ODIM;
            #pragma unroll
            for (int v = 0; v < 8; ++v) {
                const int m = v + 8 * half;
                op[m * ODIM + wv * 16 + lm] = fmaxf(ao[v], 0.0f);
            }
        }
        // next iteration's first barrier orders everything else
    }

    __syncthreads();
    // ---- final state writeback: h_fin then md_fin, appended after output ----
    float* hfin = out + (size_t)T_STEPS * BATCH * ODIM;
    for (int i = tid; i < 16 * HDIM; i += 256) {
        int m = i >> 9, k = i & 511;
        hfin[(size_t)(M0 + m) * HDIM + k] = sh_h32[m][k];
    }
    if (wv == 7) {
        float* mdfin = out + (size_t)T_STEPS * BATCH * ODIM + (size_t)BATCH * HDIM;
        #pragma unroll
        for (int v = 0; v < 8; ++v) {
            const int m = v + 8 * half;
            mdfin[(M0 + m) * MDIM + lm] = mdreg[v];
        }
    }
}

// ---------------------------------------------------------------------------
extern "C" void kernel_launch(void* const* d_in, const int* in_sizes, int n_in,
                              void* d_out, int out_size, void* d_ws, size_t ws_size,
                              hipStream_t stream) {
    const float* x      = (const float*)d_in[0];
    // d_in[1] = task_id (unused)
    const float* x2h_w  = (const float*)d_in[2];
    const float* x2h_b  = (const float*)d_in[3];
    const float* h2h_w  = (const float*)d_in[4];
    const float* h2h_b  = (const float*)d_in[5];
    const float* h2md_w = (const float*)d_in[6];
    const float* h2md_b = (const float*)d_in[7];
    const float* x2md_w = (const float*)d_in[8];
    const float* x2md_b = (const float*)d_in[9];
    const float* h2r_w  = (const float*)d_in[10];
    const float* h2r_b  = (const float*)d_in[11];
    const float* mg     = (const float*)d_in[12];

    unsigned short* ws = (unsigned short*)d_ws;
    float* out = (float*)d_out;

    prep_weights<<<(WS_TOTAL + 255) / 256, 256, 0, stream>>>(
        x2h_w, h2h_w, h2md_w, x2md_w, mg, h2r_w, ws);

    ctrnn_md_kernel<<<BATCH / 16, 256, 0, stream>>>(
        x, x2h_b, h2h_b, h2md_b, x2md_b, h2r_b, ws, out);
}
